// TorchMiniBlock_21861383537483
// MI455X (gfx1250) — matrix operand order repaired and checked by reading, not compiled, not run on hardware
//
#include <hip/hip_runtime.h>
#include <math.h>

typedef __attribute__((ext_vector_type(16))) _Float16 v16h;
typedef __attribute__((ext_vector_type(8)))  _Float16 v8h;
typedef __attribute__((ext_vector_type(8)))  float    v8f;

#define T_DIM 2048
#define D_DIM 64
#define BH_DIM 32                  // B*H
#define TQ 16
#define WAVES 8
#define QBLK (TQ * WAVES)          // 128 queries per workgroup
#define LDS_STRIDE 72              // halves per LDS row (64 + pad)

// 1/sqrt(64) * log2(e): fold softmax scale + exp->exp2 base change into Q
#define QSCALE_F 0.18033688011112042f

__device__ __forceinline__ v8f wmma_f16(v16h a, v16h b, v8f c) {
  // D = A(16x32 f16) * B(32x16 f16) + C(16x16 f32)
  return __builtin_amdgcn_wmma_f32_16x16x32_f16(
      /*neg_a=*/false, a, /*neg_b=*/false, b,
      /*c_mod=*/(short)0, c, /*reuse_a=*/false, /*reuse_b=*/false);
}

// ---------------- DPP16 butterfly reductions within 16-lane groups ----------------
// xor1: quad_perm(1,0,3,2)=0xB1 ; xor2: quad_perm(2,3,0,1)=0x4E
// xor4-equivalent (value uniform per quad): row_half_mirror=0x141 (lane^7)
// xor8-equivalent (value uniform per octet): row_mirror=0x140 (lane^15)
template <int CTRL>
__device__ __forceinline__ float dpp_f32(float v) {
  int r = __builtin_amdgcn_mov_dpp(__builtin_bit_cast(int, v), CTRL, 0xf, 0xf, true);
  return __builtin_bit_cast(float, r);
}
__device__ __forceinline__ float redmax16(float v) {
  v = fmaxf(v, dpp_f32<0xB1>(v));
  v = fmaxf(v, dpp_f32<0x4E>(v));
  v = fmaxf(v, dpp_f32<0x141>(v));
  v = fmaxf(v, dpp_f32<0x140>(v));
  return v;
}
__device__ __forceinline__ float redsum16(float v) {
  v += dpp_f32<0xB1>(v);
  v += dpp_f32<0x4E>(v);
  v += dpp_f32<0x141>(v);
  v += dpp_f32<0x140>(v);
  return v;
}

// ---------------- f32-source fragment loaders (fallback path) ----------------

__device__ __forceinline__ v16h load_A_f32(const float* __restrict__ base, int ld,
                                           int k0, int lane16, int half) {
  const float* p = base + lane16 * ld + k0 + 16 * half;
  float4 f0 = *(const float4*)(p);
  float4 f1 = *(const float4*)(p + 4);
  float4 f2 = *(const float4*)(p + 8);
  float4 f3 = *(const float4*)(p + 12);
  v16h a;
  a[0]=(_Float16)f0.x; a[1]=(_Float16)f0.y; a[2]=(_Float16)f0.z; a[3]=(_Float16)f0.w;
  a[4]=(_Float16)f1.x; a[5]=(_Float16)f1.y; a[6]=(_Float16)f1.z; a[7]=(_Float16)f1.w;
  a[8]=(_Float16)f2.x; a[9]=(_Float16)f2.y; a[10]=(_Float16)f2.z; a[11]=(_Float16)f2.w;
  a[12]=(_Float16)f3.x; a[13]=(_Float16)f3.y; a[14]=(_Float16)f3.z; a[15]=(_Float16)f3.w;
  return a;
}

__device__ __forceinline__ v16h load_B_rows_f32(const float* __restrict__ src, int ld,
                                                int row0, int d0, int lane16, int half) {
  const float* p = src + (row0 + lane16) * ld + d0 + 16 * half;
  float4 f0 = *(const float4*)(p);
  float4 f1 = *(const float4*)(p + 4);
  float4 f2 = *(const float4*)(p + 8);
  float4 f3 = *(const float4*)(p + 12);
  v16h b;
  b[0]=(_Float16)f0.x; b[1]=(_Float16)f0.y; b[2]=(_Float16)f0.z; b[3]=(_Float16)f0.w;
  b[4]=(_Float16)f1.x; b[5]=(_Float16)f1.y; b[6]=(_Float16)f1.z; b[7]=(_Float16)f1.w;
  b[8]=(_Float16)f2.x; b[9]=(_Float16)f2.y; b[10]=(_Float16)f2.z; b[11]=(_Float16)f2.w;
  b[12]=(_Float16)f3.x; b[13]=(_Float16)f3.y; b[14]=(_Float16)f3.z; b[15]=(_Float16)f3.w;
  return b;
}

__device__ __forceinline__ v16h load_B_V_f32(const float* __restrict__ v, int k0,
                                             int dn0, int lane16, int half) {
  const float* p = v + (k0 + 16 * half) * D_DIM + dn0 + lane16;
  v16h b;
  #pragma unroll
  for (int i = 0; i < 16; ++i) b[i] = (_Float16)p[i * D_DIM];
  return b;
}

// ---------------- f16-source fragment loaders (fast path) ----------------

// A-matrix (16x32 f16): lane M: K=0..15 ; lane M+16: K=16..31
__device__ __forceinline__ v16h load_A_h16(const _Float16* __restrict__ base, int ld,
                                           int k0, int lane16, int half) {
  const _Float16* p = base + lane16 * ld + k0 + 16 * half;
  v8h lo = *(const v8h*)(p);
  v8h hi = *(const v8h*)(p + 8);
  v16h a;
  #pragma unroll
  for (int i = 0; i < 8; ++i) { a[i] = lo[i]; a[8 + i] = hi[i]; }
  return a;
}

// B-matrix (32x16 f16) from 16 contiguous halves per lane (lane n: K=0..15 / n+16: K=16..31)
__device__ __forceinline__ v16h load_B16_contig(const _Float16* __restrict__ p) {
  v8h lo = *(const v8h*)(p);
  v8h hi = *(const v8h*)(p + 8);
  v16h b;
  #pragma unroll
  for (int i = 0; i < 8; ++i) { b[i] = lo[i]; b[8 + i] = hi[i]; }
  return b;
}

// A-matrix from per-wave LDS scratch (144B row stride keeps 16B alignment -> ds_load_b128)
__device__ __forceinline__ v16h load_A_lds(const _Float16* __restrict__ S,
                                           int k0, int lane16, int half) {
  const _Float16* p = S + lane16 * LDS_STRIDE + k0 + 16 * half;
  v16h a;
  #pragma unroll
  for (int i = 0; i < 8; ++i) { a[i] = p[i]; a[8 + i] = p[8 + i]; }
  return a;
}

// -------- prepass: x(f32,[BH,T,D]) -> xh(f16 row-major) + xhT(f16,[BH,D,T]); W -> Wh --------
__global__ __launch_bounds__(256, 1)
void cvt_prepass_kernel(const float* __restrict__ x, const float* __restrict__ W,
                        _Float16* __restrict__ xh, _Float16* __restrict__ xhT,
                        _Float16* __restrict__ Wh) {
  __shared__ _Float16 tile[64 * 65];
  const int tpb = T_DIM / 64;               // 32 row-tiles per (b,h)
  const int bh  = blockIdx.x / tpb;
  const int t0  = (blockIdx.x % tpb) * 64;

  const float*  src  = x   + (size_t)bh * T_DIM * D_DIM + (size_t)t0 * D_DIM;
  _Float16*     dst  = xh  + (size_t)bh * T_DIM * D_DIM + (size_t)t0 * D_DIM;
  _Float16*     dstT = xhT + (size_t)bh * T_DIM * D_DIM; // [D][T]

  #pragma unroll
  for (int i = 0; i < 16; ++i) {
    const int idx = threadIdx.x + i * 256;  // 0..4095 over a 64x64 tile
    const _Float16 h = (_Float16)src[idx];
    dst[idx] = h;
    tile[(idx & 63) * 65 + (idx >> 6)] = h; // tile[d][t]
  }
  __syncthreads();
  #pragma unroll
  for (int i = 0; i < 16; ++i) {
    const int idx = threadIdx.x + i * 256;
    const int d = idx >> 6, t = idx & 63;
    dstT[(size_t)d * T_DIM + t0 + t] = tile[d * 65 + t];
  }
  if (blockIdx.x == 0) {
    #pragma unroll
    for (int i = 0; i < 16; ++i) {
      const int idx = threadIdx.x + i * 256;
      Wh[idx] = (_Float16)W[idx];
    }
  }
}

// ---------------- one 32-key flash-attention block ----------------
template <bool F16, bool MASK>
__device__ __forceinline__ void fa_kblock(
    int kb, int q0, int lane16, int half,
    const v16h& Qa0, const v16h& Qa1,
    const float* __restrict__ xbh,
    const _Float16* __restrict__ xhbh, const _Float16* __restrict__ xhTbh,
    _Float16* __restrict__ S,
    v8f& o0, v8f& o1, v8f& o2, v8f& o3,
    float (&m)[8], float (&l)[8]) {
  // ---- S = Q K^T (log2-domain scores) : two 16x16 tiles ----
  v8f s0 = {}, s1 = {};
  if constexpr (F16) {
    s0 = wmma_f16(Qa0, load_B16_contig(xhbh + (kb      + lane16) * D_DIM +  0 + 16 * half), s0);
    s0 = wmma_f16(Qa1, load_B16_contig(xhbh + (kb      + lane16) * D_DIM + 32 + 16 * half), s0);
    s1 = wmma_f16(Qa0, load_B16_contig(xhbh + (kb + 16 + lane16) * D_DIM +  0 + 16 * half), s1);
    s1 = wmma_f16(Qa1, load_B16_contig(xhbh + (kb + 16 + lane16) * D_DIM + 32 + 16 * half), s1);
  } else {
    s0 = wmma_f16(Qa0, load_B_rows_f32(xbh, D_DIM, kb,      0,  lane16, half), s0);
    s0 = wmma_f16(Qa1, load_B_rows_f32(xbh, D_DIM, kb,      32, lane16, half), s0);
    s1 = wmma_f16(Qa0, load_B_rows_f32(xbh, D_DIM, kb + 16, 0,  lane16, half), s1);
    s1 = wmma_f16(Qa1, load_B_rows_f32(xbh, D_DIM, kb + 16, 32, lane16, half), s1);
  }

  // ---- online softmax in exp2 domain + stage P to LDS ----
  #pragma unroll
  for (int r = 0; r < 8; ++r) {
    float a0 = s0[r];
    float a1 = s1[r];
    if constexpr (MASK) {
      const int qrow = q0 + r + 8 * half;
      if (kb + lane16      > qrow) a0 = -__builtin_inff();
      if (kb + 16 + lane16 > qrow) a1 = -__builtin_inff();
    }
    float rmax = redmax16(fmaxf(a0, a1));
    float mnew = fmaxf(m[r], rmax);
    float corr = __builtin_amdgcn_exp2f(m[r] - mnew);
    float p0   = __builtin_amdgcn_exp2f(a0 - mnew);
    float p1   = __builtin_amdgcn_exp2f(a1 - mnew);
    float rsum = redsum16(p0 + p1);
    l[r] = l[r] * corr + rsum;
    m[r] = mnew;
    o0[r] *= corr; o1[r] *= corr; o2[r] *= corr; o3[r] *= corr;

    const int row = r + 8 * half;
    S[row * LDS_STRIDE + lane16]      = (_Float16)p0;
    S[row * LDS_STRIDE + 16 + lane16] = (_Float16)p1;
  }
  asm volatile("s_wait_dscnt 0" ::: "memory");

  // ---- O += P * V  (P: 16x32 A-matrix from LDS, V: four 32x16 B-tiles) ----
  const v16h Pa = load_A_lds(S, 0, lane16, half);
  if constexpr (F16) {
    o0 = wmma_f16(Pa, load_B16_contig(xhTbh + ( 0 + lane16) * T_DIM + kb + 16 * half), o0);
    o1 = wmma_f16(Pa, load_B16_contig(xhTbh + (16 + lane16) * T_DIM + kb + 16 * half), o1);
    o2 = wmma_f16(Pa, load_B16_contig(xhTbh + (32 + lane16) * T_DIM + kb + 16 * half), o2);
    o3 = wmma_f16(Pa, load_B16_contig(xhTbh + (48 + lane16) * T_DIM + kb + 16 * half), o3);
  } else {
    o0 = wmma_f16(Pa, load_B_V_f32(xbh, kb, 0,  lane16, half), o0);
    o1 = wmma_f16(Pa, load_B_V_f32(xbh, kb, 16, lane16, half), o1);
    o2 = wmma_f16(Pa, load_B_V_f32(xbh, kb, 32, lane16, half), o2);
    o3 = wmma_f16(Pa, load_B_V_f32(xbh, kb, 48, lane16, half), o3);
  }
  asm volatile("s_wait_dscnt 0" ::: "memory");
}

// ---------------- flash attention + projection ----------------
template <bool F16>
__global__ __launch_bounds__(256, 1)
void fa_causal_proj_kernel(const float* __restrict__ x, const float* __restrict__ W,
                           const _Float16* __restrict__ xh,
                           const _Float16* __restrict__ xhT,
                           const _Float16* __restrict__ Wh,
                           float* __restrict__ out) {
  __shared__ _Float16 lds_all[WAVES * 16 * LDS_STRIDE];

  const int lane   = threadIdx.x & 31;
  const int lane16 = lane & 15;
  const int half   = lane >> 4;
  const int w      = threadIdx.x >> 5;

  const int qtiles = T_DIM / QBLK;          // 16
  const int bh     = blockIdx.x / qtiles;   // 0..31
  const int qb     = blockIdx.x % qtiles;
  const int q0     = qb * QBLK + w * TQ;    // first query row of this wave

  const float*    xbh   = x + (size_t)bh * T_DIM * D_DIM;
  const _Float16* xhbh  = F16 ? (xh  + (size_t)bh * T_DIM * D_DIM) : nullptr;
  const _Float16* xhTbh = F16 ? (xhT + (size_t)bh * T_DIM * D_DIM) : nullptr;
  float*          ybh   = out + (size_t)bh * T_DIM * D_DIM;
  _Float16*       S     = lds_all + w * 16 * LDS_STRIDE;

  // Resident Q tile: two A-matrices covering d=0..31 and d=32..63,
  // pre-scaled by 1/sqrt(D) * log2(e) so scores feed exp2 directly.
  v16h Qa0, Qa1;
  if constexpr (F16) {
    const _Float16* qbase = xhbh + (size_t)q0 * D_DIM;
    Qa0 = load_A_h16(qbase, D_DIM, 0,  lane16, half);
    Qa1 = load_A_h16(qbase, D_DIM, 32, lane16, half);
  } else {
    const float* qbase = xbh + (size_t)q0 * D_DIM;
    Qa0 = load_A_f32(qbase, D_DIM, 0,  lane16, half);
    Qa1 = load_A_f32(qbase, D_DIM, 32, lane16, half);
  }
  {
    const _Float16 qs = (_Float16)QSCALE_F;
    #pragma unroll
    for (int i = 0; i < 16; ++i) { Qa0[i] *= qs; Qa1[i] *= qs; }
  }

  v8f o0 = {}, o1 = {}, o2 = {}, o3 = {};
  float m[8], l[8];
  #pragma unroll
  for (int r = 0; r < 8; ++r) { m[r] = -__builtin_inff(); l[r] = 0.0f; }

  const int kend    = q0 + TQ - 1;          // last key needed (causal)
  const int kb_last = (kend >> 5) << 5;     // the single diagonal (masked) block

  // Steady state: fully unmasked blocks (kb + 31 <= q0 guaranteed)
  for (int kb = 0; kb < kb_last; kb += 32) {
    if constexpr (F16) __builtin_prefetch(xhbh + (size_t)(kb + 32) * D_DIM, 0, 0);
    else               __builtin_prefetch(xbh  + (size_t)(kb + 32) * D_DIM, 0, 0);
    fa_kblock<F16, false>(kb, q0, lane16, half, Qa0, Qa1, xbh, xhbh, xhTbh, S,
                          o0, o1, o2, o3, m, l);
  }
  // Diagonal block with causal masking
  fa_kblock<F16, true>(kb_last, q0, lane16, half, Qa0, Qa1, xbh, xhbh, xhTbh, S,
                       o0, o1, o2, o3, m, l);

  // ---- finalize: O /= l, stage to LDS as f16 for the projection GEMM ----
  #pragma unroll
  for (int r = 0; r < 8; ++r) {
    const float inv = 1.0f / l[r];
    o0[r] *= inv; o1[r] *= inv; o2[r] *= inv; o3[r] *= inv;
    const int row = r + 8 * half;
    S[row * LDS_STRIDE +  0 + lane16] = (_Float16)o0[r];
    S[row * LDS_STRIDE + 16 + lane16] = (_Float16)o1[r];
    S[row * LDS_STRIDE + 32 + lane16] = (_Float16)o2[r];
    S[row * LDS_STRIDE + 48 + lane16] = (_Float16)o3[r];
  }
  asm volatile("s_wait_dscnt 0" ::: "memory");

  const v16h Oa0 = load_A_lds(S, 0,  lane16, half);
  const v16h Oa1 = load_A_lds(S, 32, lane16, half);

  // ---- y = O @ W^T : B[d][e] = W[e][d] -> W rows are B columns (coalesced) ----
  #pragma unroll
  for (int et = 0; et < 4; ++et) {
    v8f y = {};
    if constexpr (F16) {
      y = wmma_f16(Oa0, load_B16_contig(Wh + (et * 16 + lane16) * D_DIM +  0 + 16 * half), y);
      y = wmma_f16(Oa1, load_B16_contig(Wh + (et * 16 + lane16) * D_DIM + 32 + 16 * half), y);
    } else {
      y = wmma_f16(Oa0, load_B_rows_f32(W, D_DIM, et * 16, 0,  lane16, half), y);
      y = wmma_f16(Oa1, load_B_rows_f32(W, D_DIM, et * 16, 32, lane16, half), y);
    }
    #pragma unroll
    for (int r = 0; r < 8; ++r) {
      const int row = q0 + r + 8 * half;
      ybh[(size_t)row * D_DIM + et * 16 + lane16] = y[r];
    }
  }
}

extern "C" void kernel_launch(void* const* d_in, const int* in_sizes, int n_in,
                              void* d_out, int out_size, void* d_ws, size_t ws_size,
                              hipStream_t stream) {
  (void)in_sizes; (void)n_in; (void)out_size;
  const float* x = (const float*)d_in[0];   // [B,H,T,D] f32
  const float* W = (const float*)d_in[1];   // [D,D] f32
  float* y = (float*)d_out;                 // [B,H,T,D] f32

  const size_t XH_ELEMS = (size_t)BH_DIM * T_DIM * D_DIM;          // 4M halves
  const size_t NEED = XH_ELEMS * 2 * sizeof(_Float16)              // xh + xhT
                    + (size_t)D_DIM * D_DIM * sizeof(_Float16);    // Wh

  const int grid_fa = BH_DIM * (T_DIM / QBLK);                     // 512

  if (ws_size >= NEED) {
    _Float16* xh  = (_Float16*)d_ws;
    _Float16* xhT = xh + XH_ELEMS;
    _Float16* Wh  = xhT + XH_ELEMS;
    cvt_prepass_kernel<<<BH_DIM * (T_DIM / 64), 256, 0, stream>>>(x, W, xh, xhT, Wh);
    fa_causal_proj_kernel<true><<<grid_fa, WAVES * 32, 0, stream>>>(x, W, xh, xhT, Wh, y);
  } else {
    fa_causal_proj_kernel<false><<<grid_fa, WAVES * 32, 0, stream>>>(x, W, nullptr, nullptr, nullptr, y);
  }
}
